// _MemoryGraphWorldModel_455266533874
// MI455X (gfx1250) — matrix operand order repaired and checked by reading, not compiled, not run on hardware
//
#include <hip/hip_runtime.h>
#include <hip/hip_bf16.h>
#include <math.h>

typedef __bf16          v16bf __attribute__((ext_vector_type(16)));
typedef __bf16          v2bf  __attribute__((ext_vector_type(2)));
typedef float           v8f   __attribute__((ext_vector_type(8)));
typedef unsigned int    v8u   __attribute__((ext_vector_type(8)));

#define N_NODES 100000
#define N_PAIRS 800000
#define IN_DIM  256
#define SLOT    128
#define MSLOTS  512
#define TOKS    2048

// workspace tile offsets (in ushort/bf16 elements); each tile = 512 bf16 = 1KB
#define ENC_T_OFF  0        // enc_W   : KT=8,  NT=8   -> 64 tiles
#define MEMT_T_OFF 32768    // memory.T: KT=4,  NT=32  -> 128 tiles
#define MEM_T_OFF  98304    // memory  : KT=16, NT=8   -> 128 tiles
#define TOK_T_OFF  163840   // tok_W   : KT=4,  NT=128 -> 512 tiles

// f32 -> bf16 via native fptrunc: backend selects hardware cvt when present
__device__ __forceinline__ unsigned short f2bf(float f) {
  __bf16 h = (__bf16)f;
  return __builtin_bit_cast(unsigned short, h);
}

__device__ __forceinline__ unsigned int pk_bf16(float a, float b) {
  v2bf v = { (__bf16)a, (__bf16)b };
  return __builtin_bit_cast(unsigned int, v);
}

// inline tanh: no OCML call (avoids caller-save VGPR spills around libm)
__device__ __forceinline__ float fast_tanh(float x) {
  x = fminf(fmaxf(x, -15.0f), 15.0f);
  float e = __expf(2.0f * x);          // v_exp_f32, inline
  return (e - 1.0f) / (e + 1.0f);      // inline rcp+NR division
}

// A-operand layout inverse: value (m, kk) with kk in [0,32) -> (lane, idx)
__device__ __forceinline__ void a_pos(int kk, int m, int& lane, int& idx) {
  int hiA = kk >> 4; idx = kk & 15; lane = m + 16 * hiA;
}

__device__ __forceinline__ v8f wmma_bf16(v16bf a, v16bf b, v8f c) {
  return __builtin_amdgcn_wmma_f32_16x16x32_bf16(false, a, false, b, (short)0, c, false, false);
}

// ---------------------------------------------------------------------------
// Kernel 0: convert f32 weights -> bf16 tiles in WMMA B-operand swizzle.
// B-operand k order used here, same as A (16-bit, 32x16): lane L holds column n = L&15; per-lane
// element i (0..15) holds k = (L>>4)*16 + i within the 32-wide K chunk.
// ---------------------------------------------------------------------------
__global__ void prep_weights_kernel(const float* __restrict__ encW,
                                    const float* __restrict__ mem,
                                    const float* __restrict__ tokW,
                                    unsigned short* __restrict__ ws) {
  int t = blockIdx.x;
  int lane = threadIdx.x;
  int n0 = lane & 15, hi = lane >> 4;
  if (t < 64) {                       // enc_W [256 x 128]
    int loc = t, kt = loc >> 3, nt = loc & 7;
    for (int i = 0; i < 16; ++i) {
      int k = kt * 32 + hi * 16 + i, n = nt * 16 + n0;
      ws[ENC_T_OFF + loc * 512 + lane * 16 + i] = f2bf(encW[k * SLOT + n]);
    }
  } else if (t < 192) {               // memory.T: B[k=feat][n=slot] = mem[n][k]
    int loc = t - 64, kt = loc >> 5, nt = loc & 31;
    for (int i = 0; i < 16; ++i) {
      int k = kt * 32 + hi * 16 + i, n = nt * 16 + n0;
      ws[MEMT_T_OFF + loc * 512 + lane * 16 + i] = f2bf(mem[n * SLOT + k]);
    }
  } else if (t < 320) {               // memory [512 x 128] natural
    int loc = t - 192, kt = loc >> 3, nt = loc & 7;
    for (int i = 0; i < 16; ++i) {
      int k = kt * 32 + hi * 16 + i, n = nt * 16 + n0;
      ws[MEM_T_OFF + loc * 512 + lane * 16 + i] = f2bf(mem[k * SLOT + n]);
    }
  } else {                            // tok_W [128 x 2048]
    int loc = t - 320, kt = loc >> 7, nt = loc & 127;
    for (int i = 0; i < 16; ++i) {
      int k = kt * 32 + hi * 16 + i, n = nt * 16 + n0;
      ws[TOK_T_OFF + loc * 512 + lane * 16 + i] = f2bf(tokW[k * TOKS + n]);
    }
  }
}

// Build A-operand v16bf for one 16x32 K-chunk from a row-major f32 row.
// Lane L: m = L&15, hi = L>>4; element i -> k = base+hi*16+i
__device__ __forceinline__ v16bf load_a_tile(const float* __restrict__ arow, int kbase, int hi) {
  const float4* c0 = (const float4*)(arow + kbase + hi * 16);
  const float4* c1 = (const float4*)(arow + kbase + hi * 16 + 8);
  float4 x0 = c0[0], x1 = c0[1], y0 = c1[0], y1 = c1[1];
  v8u au;
  au[0] = pk_bf16(x0.x, x0.y); au[1] = pk_bf16(x0.z, x0.w);
  au[2] = pk_bf16(x1.x, x1.y); au[3] = pk_bf16(x1.z, x1.w);
  au[4] = pk_bf16(y0.x, y0.y); au[5] = pk_bf16(y0.z, y0.w);
  au[6] = pk_bf16(y1.x, y1.y); au[7] = pk_bf16(y1.z, y1.w);
  return __builtin_bit_cast(v16bf, au);
}

// ---------------------------------------------------------------------------
// Kernel A: fused  encoded = tanh(nf@enc_W+b);  attn = softmax(encoded@mem.T/s);
//                  node_hidden = attn@mem.   One 16-node tile per wave.
// ---------------------------------------------------------------------------
__global__ void __launch_bounds__(256, 1)
node_attention_kernel(const float* __restrict__ nf, const float* __restrict__ enc_b,
                      const unsigned short* __restrict__ ws, float* __restrict__ nh) {
  __shared__ __align__(32) unsigned short sEnc[8][2048];  // per-wave encoded, A-layout
  __shared__ __align__(32) unsigned short sB[8][512];     // per-wave bounce tile
  const int lane = threadIdx.x & 31, wave = threadIdx.x >> 5;
  const int m0 = lane & 15, hi = lane >> 4;
  const int rowBase = (blockIdx.x * 8 + wave) * 16;
  const int mrow = rowBase + m0;
  const int mc = (mrow < N_NODES) ? mrow : (N_NODES - 1);
  const float* arow = nf + (size_t)mc * IN_DIM;

  // ---- Phase 1: encoded[16x128] = tanh(A[16x256] @ enc_W[256x128] + b)
  v8f accE[8];
  for (int n = 0; n < 8; ++n) for (int r = 0; r < 8; ++r) accE[n][r] = 0.0f;
  for (int kc = 0; kc < 8; ++kc) {
    v16bf a = load_a_tile(arow, kc * 32, hi);
    for (int n = 0; n < 8; ++n) {
      v16bf b = *(const v16bf*)(ws + ENC_T_OFF + (size_t)(kc * 8 + n) * 512 + lane * 16);
      accE[n] = wmma_bf16(a, b, accE[n]);
    }
  }
  // bias + tanh, retile C-layout -> A-layout through LDS
  for (int n = 0; n < 8; ++n) {
    float bias = enc_b[n * 16 + m0];
    for (int r = 0; r < 8; ++r) {
      float v = fast_tanh(accE[n][r] + bias);
      int m = r + 8 * hi, col = n * 16 + m0;
      int kt = col >> 5, kk = col & 31, lA, idx;
      a_pos(kk, m, lA, idx);
      sEnc[wave][kt * 512 + lA * 16 + idx] = f2bf(v);
    }
  }
  __syncthreads();
  v16bf aE[4];
  for (int kt = 0; kt < 4; ++kt)
    aE[kt] = *(const v16bf*)&sEnc[wave][kt * 512 + lane * 16];

  // ---- Phase 2a: row max of logits (K=128, N=512)
  const float scale = 0.08838834764831845f;  // 1/sqrt(128)
  float maxv[8];
  for (int r = 0; r < 8; ++r) maxv[r] = -3.0e38f;
  for (int nt = 0; nt < 32; ++nt) {
    v8f c; for (int r = 0; r < 8; ++r) c[r] = 0.0f;
    for (int kt = 0; kt < 4; ++kt) {
      v16bf b = *(const v16bf*)(ws + MEMT_T_OFF + (size_t)(kt * 32 + nt) * 512 + lane * 16);
      c = wmma_bf16(aE[kt], b, c);
    }
    for (int r = 0; r < 8; ++r) maxv[r] = fmaxf(maxv[r], c[r] * scale);
  }
  for (int off = 1; off < 16; off <<= 1)
    for (int r = 0; r < 8; ++r) maxv[r] = fmaxf(maxv[r], __shfl_xor(maxv[r], off, 32));

  // ---- Phase 2b: recompute logits, exp (unnormalized), GEMM2 on the fly
  v8f accH[8];
  for (int n = 0; n < 8; ++n) for (int r = 0; r < 8; ++r) accH[n][r] = 0.0f;
  float sum[8]; for (int r = 0; r < 8; ++r) sum[r] = 0.0f;
  for (int kt2 = 0; kt2 < 16; ++kt2) {   // attention K-chunks of 32 slots
    for (int sub = 0; sub < 2; ++sub) {
      int nt = kt2 * 2 + sub;
      v8f c; for (int r = 0; r < 8; ++r) c[r] = 0.0f;
      for (int kt = 0; kt < 4; ++kt) {
        v16bf b = *(const v16bf*)(ws + MEMT_T_OFF + (size_t)(kt * 32 + nt) * 512 + lane * 16);
        c = wmma_bf16(aE[kt], b, c);
      }
      for (int r = 0; r < 8; ++r) {
        float e = __expf(c[r] * scale - maxv[r]);   // inline v_exp_f32, no call
        sum[r] += e;
        int m = r + 8 * hi, kk = sub * 16 + m0, lA, idx;
        a_pos(kk, m, lA, idx);
        sB[wave][lA * 16 + idx] = f2bf(e);
      }
    }
    __syncthreads();
    v16bf aA = *(const v16bf*)&sB[wave][lane * 16];
    for (int n = 0; n < 8; ++n) {
      v16bf b = *(const v16bf*)(ws + MEM_T_OFF + (size_t)(kt2 * 8 + n) * 512 + lane * 16);
      accH[n] = wmma_bf16(aA, b, accH[n]);
    }
    __syncthreads();
  }
  for (int off = 1; off < 16; off <<= 1)
    for (int r = 0; r < 8; ++r) sum[r] += __shfl_xor(sum[r], off, 32);
  float rinv[8];
  for (int r = 0; r < 8; ++r) rinv[r] = 1.0f / sum[r];
  // store node_hidden (half-wave rows are contiguous in n -> coalesced 64B)
  for (int n = 0; n < 8; ++n)
    for (int r = 0; r < 8; ++r) {
      int row = rowBase + r + 8 * hi;
      if (row < N_NODES) nh[(size_t)row * SLOT + n * 16 + m0] = accH[n][r] * rinv[r];
    }
}

// ---------------------------------------------------------------------------
// Kernel B: token_logits = node_hidden[100000x128] @ tok_W[128x2048] + tok_b
// 32 rows per wave (two 16-row A-tile sets) -> halves B-tile L2 traffic.
// ---------------------------------------------------------------------------
__global__ void __launch_bounds__(256, 1)
token_kernel(const float* __restrict__ nh, const float* __restrict__ tok_b,
             const unsigned short* __restrict__ ws, float* __restrict__ out) {
  const int lane = threadIdx.x & 31, wave = threadIdx.x >> 5;
  const int m0 = lane & 15, hi = lane >> 4;
  const int rowBase = (blockIdx.x * 8 + wave) * 32;      // 32 rows per wave
  const int r0 = rowBase + m0;
  const int r1 = rowBase + 16 + m0;
  const int c0r = (r0 < N_NODES) ? r0 : (N_NODES - 1);
  const int c1r = (r1 < N_NODES) ? r1 : (N_NODES - 1);
  const float* arow0 = nh + (size_t)c0r * SLOT;
  const float* arow1 = nh + (size_t)c1r * SLOT;
  v16bf aT0[4], aT1[4];
  for (int kc = 0; kc < 4; ++kc) {
    aT0[kc] = load_a_tile(arow0, kc * 32, hi);
    aT1[kc] = load_a_tile(arow1, kc * 32, hi);
  }
  for (int nt = 0; nt < 128; ++nt) {
    v8f c0, c1;
    for (int r = 0; r < 8; ++r) { c0[r] = 0.0f; c1[r] = 0.0f; }
    for (int kt = 0; kt < 4; ++kt) {
      const unsigned short* bp = ws + TOK_T_OFF + (size_t)(kt * 128 + nt) * 512 + lane * 16;
      if (nt + 1 < 128)  // prefetch next B tile (global_prefetch_b8)
        __builtin_prefetch(bp + 512, 0, 0);
      v16bf b = *(const v16bf*)bp;
      c0 = wmma_bf16(aT0[kt], b, c0);
      c1 = wmma_bf16(aT1[kt], b, c1);
    }
    float bias = tok_b[nt * 16 + m0];
    for (int r = 0; r < 8; ++r) {
      int row0 = rowBase + r + 8 * hi;
      int row1 = rowBase + 16 + r + 8 * hi;
      if (row0 < N_NODES) out[(size_t)row0 * TOKS + nt * 16 + m0] = c0[r] + bias;
      if (row1 < N_NODES) out[(size_t)row1 * TOKS + nt * 16 + m0] = c1[r] + bias;
    }
  }
}

// ---------------------------------------------------------------------------
// Kernel C: pair heads. One wave per pair; float4 per lane; shuffle reduce.
// node_hidden is 51 MB -> L2 resident, so these gathers hit L2.
// ---------------------------------------------------------------------------
__device__ __forceinline__ float dot4(float4 a, float4 b) {
  return a.x * b.x + a.y * b.y + a.z * b.z + a.w * b.w;
}

__global__ void __launch_bounds__(256)
pair_kernel(const long long* __restrict__ pidx, const float* __restrict__ nh,
            const float* __restrict__ eW, const float* __restrict__ eb,
            const float* __restrict__ gW, const float* __restrict__ gb,
            const float* __restrict__ tW, const float* __restrict__ tb,
            float* __restrict__ oE, float* __restrict__ oG, float* __restrict__ oT) {
  const int lane = threadIdx.x & 31, wave = threadIdx.x >> 5;
  const long long p = (long long)blockIdx.x * 8 + wave;
  if (p >= N_PAIRS) return;
  const int li = (int)pidx[2 * p], ri = (int)pidx[2 * p + 1];
  float4 l4 = *(const float4*)(nh + (size_t)li * SLOT + lane * 4);
  float4 r4 = *(const float4*)(nh + (size_t)ri * SLOT + lane * 4);
  float4 ewl = *(const float4*)(eW + lane * 4);
  float4 ewr = *(const float4*)(eW + SLOT + lane * 4);
  float4 gwl = *(const float4*)(gW + lane * 4);
  float4 gwr = *(const float4*)(gW + SLOT + lane * 4);
  float4 twl = *(const float4*)(tW + lane * 4);
  float4 twr = *(const float4*)(tW + SLOT + lane * 4);
  float se = dot4(l4, ewl) + dot4(r4, ewr);
  float sg = dot4(l4, gwl) + dot4(r4, gwr);
  float st = dot4(l4, twl) + dot4(r4, twr);
  for (int off = 1; off < 32; off <<= 1) {
    se += __shfl_xor(se, off, 32);
    sg += __shfl_xor(sg, off, 32);
    st += __shfl_xor(st, off, 32);
  }
  if (lane == 0) {
    oE[p] = se + eb[0];
    oG[p] = sg + gb[0];
    oT[p] = st + tb[0];
  }
}

extern "C" void kernel_launch(void* const* d_in, const int* in_sizes, int n_in,
                              void* d_out, int out_size, void* d_ws, size_t ws_size,
                              hipStream_t stream) {
  const float*     nf   = (const float*)d_in[0];
  const long long* pidx = (const long long*)d_in[1];
  const float*     encW = (const float*)d_in[2];
  const float*     encB = (const float*)d_in[3];
  const float*     mem  = (const float*)d_in[4];
  const float*     eW   = (const float*)d_in[5];
  const float*     eb   = (const float*)d_in[6];
  const float*     gW   = (const float*)d_in[7];
  const float*     gb   = (const float*)d_in[8];
  const float*     tW   = (const float*)d_in[9];
  const float*     tb   = (const float*)d_in[10];
  const float*     tokW = (const float*)d_in[11];
  const float*     tokB = (const float*)d_in[12];

  float* out = (float*)d_out;
  float* nh  = out;                                   // [100000 x 128]
  float* oE  = out + (size_t)N_NODES * SLOT;          // [800000]
  float* oG  = oE + N_PAIRS;                          // [800000]
  float* oT  = oG + N_PAIRS;                          // [800000]
  float* tok = oT + N_PAIRS;                          // [100000 x 2048]
  unsigned short* ws = (unsigned short*)d_ws;

  prep_weights_kernel<<<832, 32, 0, stream>>>(encW, mem, tokW, ws);

  const int rowBlocks = (N_NODES + 127) / 128;        // 8 waves x 16 rows per block
  node_attention_kernel<<<rowBlocks, 256, 0, stream>>>(nf, encB, ws, nh);

  const int tokBlocks = (N_NODES + 255) / 256;        // 8 waves x 32 rows per block
  token_kernel<<<tokBlocks, 256, 0, stream>>>(nh, tokB, ws, tok);

  const int pairBlocks = (N_PAIRS + 7) / 8;           // 8 waves (pairs) per block
  pair_kernel<<<pairBlocks, 256, 0, stream>>>(pidx, nh, eW, eb, gW, gb, tW, tb, oE, oG, oT);
}
